// GNN_12292196402142
// MI455X (gfx1250) — compile-verified, matrix operand order checked by reading, not run on hardware
//
#include <hip/hip_runtime.h>
#include <hip/hip_bf16.h>

// ---------------------------------------------------------------------------
// Problem constants (from the reference)
// ---------------------------------------------------------------------------
#define NUM_NODES  16384      // B*N
#define NUM_EDGES  524288     // E
#define D0         263        // C + 7
#define DOUT       256        // per-layer output dim
#define OUT_ROW    1031       // 263 + 3*256
#define KPAD1      288        // D0 padded to multiple of 32
#define KPAD23     256

typedef __attribute__((ext_vector_type(16))) __bf16 v16bf;
typedef __attribute__((ext_vector_type(8)))  float  v8f;

// ---------------------------------------------------------------------------
// Kernel 1: build x0 into d_out (cols 0..262), zero agg region (263..1030),
//           and emit bf16 padded activation matrix Xb [NUM_NODES x KPAD1].
// ---------------------------------------------------------------------------
__global__ void k_init(const float* __restrict__ pooled,
                       const float* __restrict__ rois,
                       float* __restrict__ out,
                       __bf16* __restrict__ Xb)
{
    const int n = blockIdx.x;
    const float* pf = pooled + (size_t)n * 256;
    const float* rf = rois   + (size_t)n * 7;
    float*  orow = out + (size_t)n * OUT_ROW;
    __bf16* xrow = Xb  + (size_t)n * KPAD1;

    for (int k = threadIdx.x; k < OUT_ROW; k += blockDim.x) {
        float v = 0.0f;
        if (k < 256)       v = pf[k];
        else if (k < D0)   v = rf[k - 256];
        orow[k] = v;                       // zero-fills the 3 agg slots too
    }
    for (int k = threadIdx.x; k < KPAD1; k += blockDim.x) {
        float v = 0.0f;
        if (k < 256)       v = pf[k];
        else if (k < D0)   v = rf[k - 256];
        xrow[k] = (__bf16)v;
    }
}

// ---------------------------------------------------------------------------
// Kernel 2: pack weights. W is [256 x 2d] row-major (out-major).
//   Bw[col][k] (bf16, K-contiguous per column, col = 0..511):
//     col <  256 : Wa[col][k]                      (U path)
//     col >= 256 : Wb[col-256][k] - Wa[col-256][k] (V path)
//   bias2[col]: 0 for U columns, b[col-256] for V columns.
// ---------------------------------------------------------------------------
__global__ void k_pack_w(const float* __restrict__ W,
                         const float* __restrict__ b,
                         __bf16* __restrict__ Bw,
                         float* __restrict__ bias2,
                         int d, int Kpad)
{
    const int col = blockIdx.x;           // 0..511
    for (int k = threadIdx.x; k < Kpad; k += blockDim.x) {
        float v = 0.0f;
        if (k < d) {
            if (col < 256) {
                v = W[(size_t)col * (2 * d) + k];
            } else {
                const size_t r = (size_t)(col - 256) * (2 * d);
                v = W[r + d + k] - W[r + k];
            }
        }
        Bw[(size_t)col * Kpad + k] = (__bf16)v;
    }
    if (threadIdx.x == 0) bias2[col] = (col < 256) ? 0.0f : b[col - 256];
}

// ---------------------------------------------------------------------------
// Kernel 3: WMMA GEMM  UV[16384 x 512] = Xb[16384 x KP] * Bw^T (+ bias2)
// One wave computes a 16x128 tile: 1 A fragment reused by 8 B fragments
// (8 WMMAs per 18 b128 loads), with the next A fragment software-prefetched.
// Lane layout used here (contiguous k, same for A and B): lane L -> row/col (L&15), K-base (L>>4)*16.
// ---------------------------------------------------------------------------
template <int KP>
__global__ __launch_bounds__(256)
void k_gemm(const __bf16* __restrict__ Xb,
            const __bf16* __restrict__ Bw,
            const float*  __restrict__ bias2,
            float* __restrict__ UV)
{
    const int gid   = blockIdx.x * blockDim.x + threadIdx.x;
    const int wave  = gid >> 5;
    const int lane  = threadIdx.x & 31;
    const int ntile = wave & 3;            // 4 tiles of 128 cols -> N = 512
    const int mtile = wave >> 2;           // 1024 tiles of 16 rows

    const int l15   = lane & 15;
    const int khalf = (lane >> 4) * 16;

    const __bf16* arow = Xb + (size_t)(mtile * 16 + l15) * KP + khalf;
    const __bf16* brow = Bw + (size_t)(ntile * 128 + l15) * KP + khalf;

    v8f acc[8];
#pragma unroll
    for (int j = 0; j < 8; ++j) acc[j] = (v8f){};

    v16bf a_cur = *(const v16bf*)(arow);

#pragma unroll
    for (int k = 0; k < KP; k += 32) {
        // prefetch next A fragment while the 8 WMMAs of this step execute
        v16bf a_nxt = (k + 32 < KP) ? *(const v16bf*)(arow + k + 32) : a_cur;
        v16bf bf[8];
#pragma unroll
        for (int j = 0; j < 8; ++j)
            bf[j] = *(const v16bf*)(brow + (size_t)(16 * j) * KP + k);
#pragma unroll
        for (int j = 0; j < 8; ++j)
            acc[j] = __builtin_amdgcn_wmma_f32_16x16x32_bf16(
                         false, a_cur, false, bf[j],
                         (short)0, acc[j], false, false);
        a_cur = a_nxt;
    }

    // D layout: VGPR r, lanes 0-15 -> M=r, lanes 16-31 -> M=8+r; N = lane&15.
    const int mbase = mtile * 16 + ((lane >> 4) << 3);
    const int c0    = ntile * 128 + l15;
    float bb[8];
#pragma unroll
    for (int j = 0; j < 8; ++j) bb[j] = bias2[c0 + 16 * j];
#pragma unroll
    for (int r = 0; r < 8; ++r) {
        float* orow = UV + (size_t)(mbase + r) * 512;
#pragma unroll
        for (int j = 0; j < 8; ++j)
            orow[c0 + 16 * j] = acc[j][r] + bb[j];
    }
}

// ---------------------------------------------------------------------------
// Kernel 4: edge phase. One wave per edge:
//   m = relu(U[src][c] + V[dst][c]);  out[dst][col_off + c] = max(out, m)
// float4 gathers (global_load_b128, 512B per wave-load, L2-resident) followed
// by GLOBAL_ATOMIC_MAX_NUM_F32 (no return -> STOREcnt). Exact & commutative
// => deterministic. agg region pre-zeroed; relu >= 0 matches PyG -inf->0.
// ---------------------------------------------------------------------------
__global__ __launch_bounds__(256)
void k_edge(const int* __restrict__ src,
            const int* __restrict__ dst,
            const float* __restrict__ UV,
            float* __restrict__ out,
            int col_off)
{
    const int gid  = blockIdx.x * blockDim.x + threadIdx.x;
    const int e    = gid >> 5;
    const int lane = threadIdx.x & 31;
    if (e >= NUM_EDGES) return;

    const int s = src[e];
    const int d = dst[e];
    const float4* u = (const float4*)(UV + (size_t)s * 512) + lane;       // U[src]
    const float4* v = (const float4*)(UV + (size_t)d * 512 + 256) + lane; // V[dst]
    float* o = out + (size_t)d * OUT_ROW + col_off + lane * 4;

#pragma unroll
    for (int h = 0; h < 2; ++h) {                 // 2 x (32 lanes x float4) = 256
        const float4 uu = u[h * 32];
        const float4 vv = v[h * 32];
        float m0 = fmaxf(uu.x + vv.x, 0.0f);
        float m1 = fmaxf(uu.y + vv.y, 0.0f);
        float m2 = fmaxf(uu.z + vv.z, 0.0f);
        float m3 = fmaxf(uu.w + vv.w, 0.0f);
        float* p = o + h * 128;
        asm volatile("global_atomic_max_num_f32 %0, %1, off" :: "v"(p + 0), "v"(m0) : "memory");
        asm volatile("global_atomic_max_num_f32 %0, %1, off" :: "v"(p + 1), "v"(m1) : "memory");
        asm volatile("global_atomic_max_num_f32 %0, %1, off" :: "v"(p + 2), "v"(m2) : "memory");
        asm volatile("global_atomic_max_num_f32 %0, %1, off" :: "v"(p + 3), "v"(m3) : "memory");
    }
}

// ---------------------------------------------------------------------------
// Kernel 5: re-pack a layer's aggregated output (fp32, strided in d_out)
//           into the bf16 activation matrix for the next layer's GEMM.
// ---------------------------------------------------------------------------
__global__ void k_pack_x(const float* __restrict__ out,
                         __bf16* __restrict__ Xb,
                         int col_off)
{
    const int n = blockIdx.x;
    const float* orow = out + (size_t)n * OUT_ROW + col_off;
    __bf16* xrow = Xb + (size_t)n * KPAD23;
    for (int k = threadIdx.x; k < KPAD23; k += blockDim.x)
        xrow[k] = (__bf16)orow[k];
}

// ---------------------------------------------------------------------------
// Host-side orchestration
// ---------------------------------------------------------------------------
extern "C" void kernel_launch(void* const* d_in, const int* in_sizes, int n_in,
                              void* d_out, int out_size, void* d_ws, size_t ws_size,
                              hipStream_t stream)
{
    const float* rois   = (const float*)d_in[0];
    const float* pooled = (const float*)d_in[1];
    const int*   eidx   = (const int*)d_in[2];
    const float* W1 = (const float*)d_in[3]; const float* b1 = (const float*)d_in[4];
    const float* W2 = (const float*)d_in[5]; const float* b2 = (const float*)d_in[6];
    const float* W3 = (const float*)d_in[7]; const float* b3 = (const float*)d_in[8];
    float* out = (float*)d_out;

    const int* src = eidx;              // edge_index[0, :]
    const int* dst = eidx + NUM_EDGES;  // edge_index[1, :]

    // Workspace layout
    char* ws = (char*)d_ws;
    const size_t SZ_XB = (size_t)NUM_NODES * KPAD1 * sizeof(__bf16);   //  9.4 MB
    const size_t SZ_UV = (size_t)NUM_NODES * 512 * sizeof(float);      // 33.6 MB
    const size_t SZ_BW = (size_t)512 * KPAD1 * sizeof(__bf16);         //  0.3 MB
    __bf16* Xb    = (__bf16*)(ws);
    float*  UV    = (float*)(ws + SZ_XB);
    __bf16* Bw    = (__bf16*)(ws + SZ_XB + SZ_UV);
    float*  bias2 = (float*)(ws + SZ_XB + SZ_UV + SZ_BW);

    const int GEMM_BLOCKS = (NUM_NODES / 16) * 4 * 32 / 256;  // 512
    const int EDGE_BLOCKS = NUM_EDGES * 32 / 256;             // 65536

    // x0 + zeroed agg region + bf16 activations
    k_init<<<NUM_NODES, 256, 0, stream>>>(pooled, rois, out, Xb);

    // ---- Layer 1 (K = 263 -> 288) ----
    k_pack_w<<<512, 256, 0, stream>>>(W1, b1, Bw, bias2, D0, KPAD1);
    k_gemm<KPAD1><<<GEMM_BLOCKS, 256, 0, stream>>>(Xb, Bw, bias2, UV);
    k_edge<<<EDGE_BLOCKS, 256, 0, stream>>>(src, dst, UV, out, D0);

    // ---- Layer 2 (K = 256) ----
    k_pack_x<<<NUM_NODES, 256, 0, stream>>>(out, Xb, D0);
    k_pack_w<<<512, 256, 0, stream>>>(W2, b2, Bw, bias2, DOUT, KPAD23);
    k_gemm<KPAD23><<<GEMM_BLOCKS, 256, 0, stream>>>(Xb, Bw, bias2, UV);
    k_edge<<<EDGE_BLOCKS, 256, 0, stream>>>(src, dst, UV, out, D0 + DOUT);

    // ---- Layer 3 (K = 256) ----
    k_pack_x<<<NUM_NODES, 256, 0, stream>>>(out, Xb, D0 + DOUT);
    k_pack_w<<<512, 256, 0, stream>>>(W3, b3, Bw, bias2, DOUT, KPAD23);
    k_gemm<KPAD23><<<GEMM_BLOCKS, 256, 0, stream>>>(Xb, Bw, bias2, UV);
    k_edge<<<EDGE_BLOCKS, 256, 0, stream>>>(src, dst, UV, out, D0 + 2 * DOUT);
}
